// VoxelSetAbstraction_81982335746525
// MI455X (gfx1250) — compile-verified, matrix operand order checked by reading, not run on hardware
//
#include <hip/hip_runtime.h>
#include <hip/hip_bf16.h>

// ---------------- problem constants (match reference setup) ----------------
#define BB    2
#define NPT   16384
#define NKP   2048
#define NS    16
#define CBEV  256
#define HBEV  200
#define WBEV  176
#define PFW   288        // 256 bev + 16 + 16
#define FOUT  128
#define EPSBN 1e-5f

typedef __attribute__((ext_vector_type(16))) _Float16 v16h;
typedef __attribute__((ext_vector_type(8)))  float    v8f;

// ---------------- workspace layout (float offsets) ----------------
#define OFF_DISTS 0
#define OFF_KP    32768
#define OFF_H0A   45056
#define OFF_H0B   (OFF_H0A + 1048576)
#define OFF_H1A   (OFF_H0B + 1048576)
#define OFF_H1B   (OFF_H1A + 1048576)
#define OFF_PF    (OFF_H1B + 1048576)
#define OFF_FPRE  (OFF_PF  + (BB*NKP*PFW))
#define OFF_STATS (OFF_FPRE + (BB*NKP*FOUT))
#define OFF_BFRAG (OFF_STATS + 1024)   // 8 ntiles * 9 ksteps * 32 lanes * 16 halves
                                       // = 36864 _Float16 = 18432 floats

// ============================================================================
// 1) Farthest-point sampling: one 1024-thread block per batch. Sequential
//    2047-iteration loop; each iter: strided min-update + LDS argmax tree.
// ============================================================================
__global__ __launch_bounds__(1024) void fps_kernel(const float* __restrict__ points,
                                                   float* __restrict__ dists,
                                                   float* __restrict__ kp,
                                                   float* __restrict__ out_coords) {
    const int b = blockIdx.x;
    const int tid = threadIdx.x;
    const float* P = points + (size_t)b * NPT * 4;
    float* dd = dists + (size_t)b * NPT;
    float* kpb = kp + (size_t)b * NKP * 3;
    float* oc  = out_coords + (size_t)b * NKP * 4;

    __shared__ float rv[1024];
    __shared__ int   ri[1024];
    __shared__ int   s_last;

    for (int i = tid; i < NPT; i += 1024) dd[i] = 1e10f;
    if (tid == 0) {
        kpb[0] = P[0]; kpb[1] = P[1]; kpb[2] = P[2];
        oc[0] = (float)b; oc[1] = P[0]; oc[2] = P[1]; oc[3] = P[2];
    }
    __syncthreads();

    int last = 0;
    for (int k = 1; k < NKP; ++k) {
        const float lx = P[last * 4 + 0];
        const float ly = P[last * 4 + 1];
        const float lz = P[last * 4 + 2];
        float best = -1.0f; int bi = 0;
        for (int i = tid; i < NPT; i += 1024) {
            const float dx = P[i * 4 + 0] - lx;
            const float dy = P[i * 4 + 1] - ly;
            const float dz = P[i * 4 + 2] - lz;
            const float d  = dx * dx + dy * dy + dz * dz;
            const float dm = fminf(dd[i], d);
            dd[i] = dm;
            if (dm > best) { best = dm; bi = i; }
        }
        rv[tid] = best; ri[tid] = bi;
        __syncthreads();
        for (int s = 512; s > 0; s >>= 1) {
            if (tid < s) {
                const float v2 = rv[tid + s]; const int i2 = ri[tid + s];
                if (v2 > rv[tid] || (v2 == rv[tid] && i2 < ri[tid])) { rv[tid] = v2; ri[tid] = i2; }
            }
            __syncthreads();
        }
        if (tid == 0) {
            const int nxt = ri[0];
            s_last = nxt;
            kpb[k * 3 + 0] = P[nxt * 4 + 0];
            kpb[k * 3 + 1] = P[nxt * 4 + 1];
            kpb[k * 3 + 2] = P[nxt * 4 + 2];
            oc[k * 4 + 0] = (float)b;
            oc[k * 4 + 1] = P[nxt * 4 + 0];
            oc[k * 4 + 2] = P[nxt * 4 + 1];
            oc[k * 4 + 3] = P[nxt * 4 + 2];
        }
        __syncthreads();
        last = s_last;
    }
}

// ============================================================================
// 2) Bilinear BEV sampling -> pf[:, 0:256]
// ============================================================================
__global__ __launch_bounds__(256) void bilinear_kernel(const float* __restrict__ bev,
                                                       const float* __restrict__ kp,
                                                       float* __restrict__ pf,
                                                       const int* __restrict__ stride_ptr) {
    const int bk = blockIdx.x;
    const int b = bk / NKP;
    const int c = threadIdx.x;
    const float st = (float)(*stride_ptr);
    const float kx = kp[(size_t)bk * 3 + 0];
    const float ky = kp[(size_t)bk * 3 + 1];
    const float xi = (kx - 0.0f) / 0.05f / st;
    const float yi = (ky - (-40.0f)) / 0.05f / st;
    const int xf = (int)floorf(xi);
    const int yf = (int)floorf(yi);
    const int x0 = min(max(xf, 0), WBEV - 1);
    const int x1 = min(max(xf + 1, 0), WBEV - 1);
    const int y0 = min(max(yf, 0), HBEV - 1);
    const int y1 = min(max(yf + 1, 0), HBEV - 1);
    const float x0f = (float)x0, x1f = (float)x1;
    const float y0f = (float)y0, y1f = (float)y1;
    const float wa = (x1f - xi) * (y1f - yi);
    const float wb = (x1f - xi) * (yi - y0f);
    const float wc = (xi - x0f) * (y1f - yi);
    const float wd = (xi - x0f) * (yi - y0f);
    const size_t base = ((size_t)b * CBEV + c) * HBEV;
    const float Ia = bev[(base + y0) * WBEV + x0];
    const float Ib = bev[(base + y1) * WBEV + x0];
    const float Ic = bev[(base + y0) * WBEV + x1];
    const float Id = bev[(base + y1) * WBEV + x1];
    pf[(size_t)bk * PFW + c] = Ia * wa + Ib * wb + Ic * wc + Id * wd;
}

// ============================================================================
// 3) Radius grouping (first-16-by-index semantics of top_k(-keys)) for both
//    radii in one ordered ballot scan, then the 4->16 layer-0 pre-activation.
//    One wave (32 threads) per keypoint.
// ============================================================================
__global__ __launch_bounds__(32) void group_mlp0_kernel(const float* __restrict__ points,
                                                        const float* __restrict__ kp,
                                                        const float* __restrict__ w0a,
                                                        const float* __restrict__ w0b,
                                                        float* __restrict__ h0a,
                                                        float* __restrict__ h0b) {
    const int bk = blockIdx.x;
    const int b = bk / NKP;
    const int lane = threadIdx.x;
    const float* P = points + (size_t)b * NPT * 4;
    const float kx = kp[(size_t)bk * 3 + 0];
    const float ky = kp[(size_t)bk * 3 + 1];
    const float kz = kp[(size_t)bk * 3 + 2];

    __shared__ int idxA[NS], idxB[NS];
    __shared__ int cA, cB;
    if (lane == 0) { cA = 0; cB = 0; }
    __syncthreads();

    int base = 0;
    while (base < NPT) {
        __syncthreads();
        if (cA >= NS && cB >= NS) break;   // uniform over the wave
        const int i = base + lane;
        const float dx = P[i * 4 + 0] - kx;
        const float dy = P[i * 4 + 1] - ky;
        const float dz = P[i * 4 + 2] - kz;
        const float d  = dx * dx + dy * dy + dz * dz;
        unsigned mA = (unsigned)__ballot(d < 0.16f);   // r=0.4
        unsigned mB = (unsigned)__ballot(d < 0.64f);   // r=0.8
        if (lane == 0) {
            while (mA && cA < NS) { idxA[cA++] = base + __builtin_ctz(mA); mA &= mA - 1; }
            while (mB && cB < NS) { idxB[cB++] = base + __builtin_ctz(mB); mB &= mB - 1; }
        }
        base += 32;
    }
    __syncthreads();
    const int nA = cA, nB = cB;
    if (lane == 0) {
        for (int t = nA; t < NS; ++t) idxA[t] = (nA > 0) ? idxA[0] : 0;
        for (int t = nB; t < NS; ++t) idxB[t] = (nB > 0) ? idxB[0] : 0;
    }
    __syncthreads();
    const bool emptyA = (nA == 0), emptyB = (nB == 0);

    for (int t = lane; t < NS * 16; t += 32) {
        const int s = t >> 4, c = t & 15;
        {
            const int id = idxA[s];
            const float gx = P[id * 4 + 0] - kx;
            const float gy = P[id * 4 + 1] - ky;
            const float gz = P[id * 4 + 2] - kz;
            const float gf = P[id * 4 + 3];
            const float h = gx * w0a[0 * 16 + c] + gy * w0a[1 * 16 + c] +
                            gz * w0a[2 * 16 + c] + gf * w0a[3 * 16 + c];
            h0a[((size_t)bk * NS + s) * 16 + c] = emptyA ? 0.0f : h;
        }
        {
            const int id = idxB[s];
            const float gx = P[id * 4 + 0] - kx;
            const float gy = P[id * 4 + 1] - ky;
            const float gz = P[id * 4 + 2] - kz;
            const float gf = P[id * 4 + 3];
            const float h = gx * w0b[0 * 16 + c] + gy * w0b[1 * 16 + c] +
                            gz * w0b[2 * 16 + c] + gf * w0b[3 * 16 + c];
            h0b[((size_t)bk * NS + s) * 16 + c] = emptyB ? 0.0f : h;
        }
    }
}

// ============================================================================
// 4) Per-channel mean/var (population) over an [M, C] tensor. grid = C.
// ============================================================================
__global__ __launch_bounds__(256) void stats_kernel(const float* __restrict__ x,
                                                    int M, int C,
                                                    float* __restrict__ mean,
                                                    float* __restrict__ var) {
    const int c = blockIdx.x;
    const int tid = threadIdx.x;
    __shared__ float ssum[256], ssq[256];
    float s = 0.0f, q = 0.0f;
    for (int r = tid; r < M; r += 256) {
        const float v = x[(size_t)r * C + c];
        s += v; q += v * v;
    }
    ssum[tid] = s; ssq[tid] = q;
    __syncthreads();
    for (int o = 128; o > 0; o >>= 1) {
        if (tid < o) { ssum[tid] += ssum[tid + o]; ssq[tid] += ssq[tid + o]; }
        __syncthreads();
    }
    if (tid == 0) {
        const float m = ssum[0] / (float)M;
        mean[c] = m;
        var[c]  = ssq[0] / (float)M - m * m;
    }
}

// ============================================================================
// 5) Fused BN-ReLU + [Mx16]@[16x16] via v_wmma_f32_16x16x32_f16 (K padded).
//    8 waves/block, one 16-row tile per wave.
// ============================================================================
__global__ __launch_bounds__(256) void bn_matmul16_kernel(const float* __restrict__ h0pre,
                                                          const float* __restrict__ mean,
                                                          const float* __restrict__ var,
                                                          const float* __restrict__ gam,
                                                          const float* __restrict__ bet,
                                                          const float* __restrict__ w1,
                                                          float* __restrict__ h1pre) {
    const int wave = threadIdx.x >> 5;
    const int lane = threadIdx.x & 31;
    const int tile = blockIdx.x * 8 + wave;
    const int rowbase = tile * 16;
    const int n  = lane & 15;
    const int hi = lane >> 4;

    // B fragment from w1 (K=16 real, padded to 32 with zeros)
    v16h bm{};
#pragma unroll
    for (int j = 0; j < 8; ++j) {
        const int kk = hi * 8 + j;                // lanes0-15: K0..7 ; lanes16-31: K8..15
        bm[j]     = (_Float16)w1[kk * 16 + n];
        bm[8 + j] = (_Float16)0.0f;               // K16..31 pad
    }

    // A fragment: BN-ReLU applied on the fly
    v16h a{};
    const int m = lane & 15;
    const float* row = h0pre + (size_t)(rowbase + m) * 16;
#pragma unroll
    for (int j = 0; j < 8; ++j) {
        const int c = hi * 8 + j;
        const float xv = row[c];
        const float xn = (xv - mean[c]) * rsqrtf(var[c] + EPSBN) * gam[c] + bet[c];
        a[j]     = (_Float16)fmaxf(xn, 0.0f);
        a[8 + j] = (_Float16)0.0f;
    }

    v8f acc = {};
    acc = __builtin_amdgcn_wmma_f32_16x16x32_f16(false, a, false, bm, (short)0, acc, false, false);

    const int mofs = hi * 8;
#pragma unroll
    for (int r = 0; r < 8; ++r)
        h1pre[(size_t)(rowbase + mofs + r) * 16 + n] = acc[r];
}

// ============================================================================
// 6) BN-ReLU + max over NSAMPLE -> pf[:, 256 + scale_off : +16]
// ============================================================================
__global__ __launch_bounds__(256) void bn_maxpool_kernel(const float* __restrict__ h1pre,
                                                         const float* __restrict__ mean,
                                                         const float* __restrict__ var,
                                                         const float* __restrict__ gam,
                                                         const float* __restrict__ bet,
                                                         float* __restrict__ pf,
                                                         int scale_off) {
    const int bk = blockIdx.x;
    const int t = threadIdx.x;
    const int s = t >> 4, c = t & 15;
    __shared__ float vals[256];
    const float xv = h1pre[((size_t)bk * NS + s) * 16 + c];
    const float xn = (xv - mean[c]) * rsqrtf(var[c] + EPSBN) * gam[c] + bet[c];
    vals[t] = fmaxf(xn, 0.0f);
    __syncthreads();
    for (int o = 128; o >= 16; o >>= 1) {
        if (t < o) vals[t] = fmaxf(vals[t], vals[t + o]);
        __syncthreads();
    }
    if (t < 16) pf[(size_t)bk * PFW + 256 + scale_off + t] = vals[t];
}

// ============================================================================
// 7a) Pre-pack fusion_w into WMMA B-fragment order (f16).
//     Layout: bfrag[((nt*9 + kk)*32 + lane)*16 + j]  -> one contiguous 32B
//     v16h per lane per K-step in the GEMM.
// ============================================================================
__global__ __launch_bounds__(32) void pack_fusion_w_kernel(const float* __restrict__ fw,
                                                           _Float16* __restrict__ bh) {
    const int blk = blockIdx.x;          // 0..71 : nt = blk/9, kk = blk%9
    const int nt = blk / 9, kk = blk % 9;
    const int lane = threadIdx.x;
    const int n  = lane & 15;
    const int hi = lane >> 4;
    const int kb = kk * 32;
    const int gcol = nt * 16 + n;
    _Float16* dst = bh + ((size_t)blk * 32 + lane) * 16;
#pragma unroll
    for (int j = 0; j < 8; ++j) {
        const int k0 = kb + hi * 8 + j;        // lanes0-15: K{0..7} ; 16-31: K{8..15}
        const int k1 = kb + 16 + hi * 8 + j;   // lanes0-15: K{16..23}; 16-31: K{24..31}
        dst[j]     = (_Float16)fw[(size_t)k0 * FOUT + gcol];
        dst[8 + j] = (_Float16)fw[(size_t)k1 * FOUT + gcol];
    }
}

// ============================================================================
// 7b) Fusion GEMM: [4096 x 288] @ [288 x 128] via WMMA f16, fully unrolled
//     K-loop of 9. A tile staged/converted in LDS; B from the packed table.
// ============================================================================
__global__ __launch_bounds__(256) void fusion_gemm_kernel(const float* __restrict__ pf,
                                                          const _Float16* __restrict__ bh,
                                                          float* __restrict__ fpre) {
    __shared__ _Float16 Ah[16 * PFW];
    const int mt = blockIdx.x;
    const int tid = threadIdx.x;
    const float* Arow = pf + (size_t)mt * 16 * PFW;
    __builtin_prefetch(Arow, 0, 1);               // global_prefetch_b8
    for (int i = tid; i < 16 * PFW; i += 256) Ah[i] = (_Float16)Arow[i];
    __syncthreads();

    const int wave = tid >> 5;                    // = N tile (0..7)
    const int lane = tid & 31;
    const int n  = lane & 15;
    const int hi = lane >> 4;
    const int m  = lane & 15;

    v8f acc = {};
#pragma unroll
    for (int kk = 0; kk < 9; ++kk) {
        const int kb = kk * 32;
        v16h a{};
        const _Float16* ar = Ah + m * PFW + kb;
#pragma unroll
        for (int j = 0; j < 8; ++j) {
            a[j]     = ar[hi * 8 + j];            // K kb+{0..7} / kb+{8..15}
            a[8 + j] = ar[16 + hi * 8 + j];       // K kb+{16..23} / kb+{24..31}
        }
        const v16h bm = *(const v16h*)(bh + (((size_t)wave * 9 + kk) * 32 + lane) * 16);
        acc = __builtin_amdgcn_wmma_f32_16x16x32_f16(false, a, false, bm, (short)0, acc, false, false);
    }
    const int mofs = hi * 8;
#pragma unroll
    for (int r = 0; r < 8; ++r)
        fpre[(size_t)(mt * 16 + mofs + r) * FOUT + wave * 16 + n] = acc[r];
}

// ============================================================================
// 8) Final BN-ReLU -> d_out fused block
// ============================================================================
__global__ __launch_bounds__(256) void bn_apply_kernel(const float* __restrict__ fpre,
                                                       const float* __restrict__ mean,
                                                       const float* __restrict__ var,
                                                       const float* __restrict__ gam,
                                                       const float* __restrict__ bet,
                                                       float* __restrict__ out) {
    const size_t i = (size_t)blockIdx.x * 256 + threadIdx.x;
    const int c = (int)(i & (FOUT - 1));
    const float xv = fpre[i];
    const float xn = (xv - mean[c]) * rsqrtf(var[c] + EPSBN) * gam[c] + bet[c];
    out[i] = fmaxf(xn, 0.0f);
}

// ============================================================================
extern "C" void kernel_launch(void* const* d_in, const int* in_sizes, int n_in,
                              void* d_out, int out_size, void* d_ws, size_t ws_size,
                              hipStream_t stream) {
    const float* points   = (const float*)d_in[0];
    const float* bev      = (const float*)d_in[1];
    const float* w0_a     = (const float*)d_in[2];
    const float* g0_a     = (const float*)d_in[3];
    const float* b0_a     = (const float*)d_in[4];
    const float* w1_a     = (const float*)d_in[5];
    const float* g1_a     = (const float*)d_in[6];
    const float* b1_a     = (const float*)d_in[7];
    const float* w0_b     = (const float*)d_in[8];
    const float* g0_b     = (const float*)d_in[9];
    const float* b0_b     = (const float*)d_in[10];
    const float* w1_b     = (const float*)d_in[11];
    const float* g1_b     = (const float*)d_in[12];
    const float* b1_b     = (const float*)d_in[13];
    const float* fusion_w = (const float*)d_in[14];
    const float* fusion_g = (const float*)d_in[15];
    const float* fusion_b = (const float*)d_in[16];
    const int*   stride_p = (const int*)d_in[17];

    float* ws    = (float*)d_ws;
    float* dists = ws + OFF_DISTS;
    float* kp    = ws + OFF_KP;
    float* h0a   = ws + OFF_H0A;
    float* h0b   = ws + OFF_H0B;
    float* h1a   = ws + OFF_H1A;
    float* h1b   = ws + OFF_H1B;
    float* pf    = ws + OFF_PF;
    float* fpre  = ws + OFF_FPRE;
    float* st    = ws + OFF_STATS;
    _Float16* bfrag = (_Float16*)(ws + OFF_BFRAG);
    // stats layout: [0]m0a [16]v0a [32]m1a [48]v1a [64]m0b [80]v0b [96]m1b [112]v1b
    //               [128..255]m_fus [256..383]v_fus

    float* out_fused  = (float*)d_out;
    float* out_coords = (float*)d_out + (size_t)BB * NKP * FOUT;

    const int BK = BB * NKP;           // 4096
    const int M16 = BK * NS;           // 65536 rows for the 16-ch layers

    fps_kernel<<<BB, 1024, 0, stream>>>(points, dists, kp, out_coords);
    pack_fusion_w_kernel<<<72, 32, 0, stream>>>(fusion_w, bfrag);
    bilinear_kernel<<<BK, 256, 0, stream>>>(bev, kp, pf, stride_p);
    group_mlp0_kernel<<<BK, 32, 0, stream>>>(points, kp, w0_a, w0_b, h0a, h0b);

    // scale A
    stats_kernel<<<16, 256, 0, stream>>>(h0a, M16, 16, st + 0,  st + 16);
    bn_matmul16_kernel<<<M16 / (16 * 8), 256, 0, stream>>>(h0a, st + 0, st + 16, g0_a, b0_a, w1_a, h1a);
    stats_kernel<<<16, 256, 0, stream>>>(h1a, M16, 16, st + 32, st + 48);
    bn_maxpool_kernel<<<BK, 256, 0, stream>>>(h1a, st + 32, st + 48, g1_a, b1_a, pf, 0);

    // scale B
    stats_kernel<<<16, 256, 0, stream>>>(h0b, M16, 16, st + 64, st + 80);
    bn_matmul16_kernel<<<M16 / (16 * 8), 256, 0, stream>>>(h0b, st + 64, st + 80, g0_b, b0_b, w1_b, h1b);
    stats_kernel<<<16, 256, 0, stream>>>(h1b, M16, 16, st + 96, st + 112);
    bn_maxpool_kernel<<<BK, 256, 0, stream>>>(h1b, st + 96, st + 112, g1_b, b1_b, pf, 16);

    // fusion
    fusion_gemm_kernel<<<BK / 16, 256, 0, stream>>>(pf, bfrag, fpre);
    stats_kernel<<<FOUT, 256, 0, stream>>>(fpre, BK, FOUT, st + 128, st + 256);
    bn_apply_kernel<<<(BK * FOUT) / 256, 256, 0, stream>>>(fpre, st + 128, st + 256, fusion_g, fusion_b, out_fused);
}
